// FractionalNeuralNetwork_54786602828150
// MI455X (gfx1250) — compile-verified, matrix operand order checked by reading, not run on hardware
//
#include <hip/hip_runtime.h>
#include <cmath>

#define BM 128
#define BN 256
#define BK 32
#define BKPAD 40            // f16 elems per LDS row: 64B data + 16B pad = 80B stride (16B aligned)
#define NTHREADS 256        // 8 wave32s: 2 along M x 4 along N, 64x64 tile per wave
#define A_TILE_ELEMS (BM * BKPAD)
#define B_TILE_ELEMS (BN * BKPAD)
#define A_TILE_BYTES (A_TILE_ELEMS * 2)  // 10240
#define B_TILE_BYTES (B_TILE_ELEMS * 2)  // 20480

typedef __attribute__((ext_vector_type(16))) _Float16 v16h;
typedef __attribute__((ext_vector_type(8)))  _Float16 v8h;
typedef __attribute__((ext_vector_type(4)))  _Float16 f16x4;
typedef __attribute__((ext_vector_type(8)))  float    v8f;
typedef __attribute__((ext_vector_type(4)))  float    f32x4;

__device__ __forceinline__ v16h ld_frag(const _Float16* p) {
    v8h lo = *(const v8h*)(p);
    v8h hi = *(const v8h*)(p + 8);
    return __builtin_shufflevector(lo, hi, 0,1,2,3,4,5,6,7,8,9,10,11,12,13,14,15);
}

// LDS byte address of a generic pointer to __shared__ (ISA 10.2: low 32 bits of the address)
__device__ __forceinline__ uint32_t lds_addr32(const void* p) {
    return (uint32_t)(uintptr_t)p;
}

// Async DMA of a ROWSx32 f16 tile (global, 64B rows) into LDS (80B row stride).
// ROWS*4 16B chunks over 256 threads -> ROWS/64 instructions/thread. Tracked with ASYNCcnt.
template<int ROWS>
__device__ __forceinline__ void stage_async(const _Float16* gbase, int ldk_bytes, int k0_bytes,
                                            uint32_t lds_base, int tid) {
    #pragma unroll
    for (int i = 0; i < ROWS / 64; ++i) {
        int c = tid + i * NTHREADS;     // chunk id: 4 per row
        int r = c >> 2;
        int o = (c & 3) * 16;
        uint32_t voff = (uint32_t)(r * ldk_bytes + k0_bytes + o);
        uint32_t ldst = lds_base + (uint32_t)(r * (BKPAD * 2) + o);
        asm volatile("global_load_async_to_lds_b128 %0, %1, %2"
                     :: "v"(ldst), "v"(voff), "s"(gbase) : "memory");
    }
}
// 6 async ops per stage (A:2 + B:4). Async ops complete in order, so waiting to <=6
// with 12 outstanding guarantees the two-steps-old stage has fully landed.
__device__ __forceinline__ void wait_async_le6() {
    asm volatile("s_wait_asynccnt 0x6" ::: "memory");
}
__device__ __forceinline__ void wait_async_le0() {
    asm volatile("s_wait_asynccnt 0x0" ::: "memory");
}

// C[M,N] = act( scale * (A[M,K]f16 x B[N,K]f16^T) + bias )
template<bool RELU, bool HAS_BIAS, bool OUT_F16, bool TRIANG>
__global__ __launch_bounds__(NTHREADS)
void gemm_f16_kernel(const _Float16* __restrict__ A,
                     const _Float16* __restrict__ B,
                     const float* __restrict__ bias,
                     void* __restrict__ Cout,
                     int M, int N, int K, float scale) {
    __shared__ __align__(16) _Float16 sA[3][A_TILE_ELEMS];
    __shared__ __align__(16) _Float16 sB[3][B_TILE_ELEMS];

    const int tid   = threadIdx.x;
    const int lane  = tid & 31;
    const int wid   = tid >> 5;
    const int wMoff = (wid & 1) * 64;    // 2 waves along M, 64 rows
    const int wNoff = (wid >> 1) * 64;   // 4 waves along N, 64 cols
    const int lrow  = lane & 15;
    const int khalf = lane >> 4;         // lanes 16-31 hold K 16..31

    const _Float16* gA = A + (size_t)blockIdx.y * BM * K;
    const _Float16* gB = B + (size_t)blockIdx.x * BN * K;
    const int ldkb = K * 2;

    v8f acc[4][4];
    #pragma unroll
    for (int mi = 0; mi < 4; ++mi)
        #pragma unroll
        for (int ni = 0; ni < 4; ++ni)
            acc[mi][ni] = (v8f)(0.0f);

    const uint32_t sAb = lds_addr32(&sA[0][0]);
    const uint32_t sBb = lds_addr32(&sB[0][0]);

    // one K-step of compute from one LDS buffer: 8 frag loads -> 16 WMMAs
    auto compute = [&](int cur) {
        const _Float16* As = &sA[0][0] + cur * A_TILE_ELEMS;
        const _Float16* Bs = &sB[0][0] + cur * B_TILE_ELEMS;
        v16h af[4], bf[4];
        #pragma unroll
        for (int mi = 0; mi < 4; ++mi)
            af[mi] = ld_frag(&As[(wMoff + mi * 16 + lrow) * BKPAD + khalf * 16]);
        #pragma unroll
        for (int ni = 0; ni < 4; ++ni)
            bf[ni] = ld_frag(&Bs[(wNoff + ni * 16 + lrow) * BKPAD + khalf * 16]);
        #pragma unroll
        for (int mi = 0; mi < 4; ++mi)
            #pragma unroll
            for (int ni = 0; ni < 4; ++ni)
                acc[mi][ni] = __builtin_amdgcn_wmma_f32_16x16x32_f16(
                    false, af[mi], false, bf[ni], (short)0, acc[mi][ni], false, false);
    };

    int KT = K / BK;
    if constexpr (TRIANG) {
        // Bt[n][k] = 0 for k > n: K-tiles past this block's last column are all-zero.
        // BN/BK == 8, so KT >= 8 and the limit is always even.
        const int lim = (blockIdx.x + 1) * (BN / BK);
        KT = (lim < KT) ? lim : KT;
    }

    // depth-2 async pipeline prologue: stages 0 and 1 in flight, wait for stage 0 only
    stage_async<BM>(gA, ldkb, 0, sAb, tid);
    stage_async<BN>(gB, ldkb, 0, sBb, tid);
    stage_async<BM>(gA, ldkb, BK * 2, sAb + A_TILE_BYTES, tid);
    stage_async<BN>(gB, ldkb, BK * 2, sBb + B_TILE_BYTES, tid);
    wait_async_le6();
    __syncthreads();

    // steady state (KT >= 8 always): unconditional stage -> compute -> wait<=6 -> barrier
    int cur = 0, nx2 = 2;     // buffer holding kt, buffer that receives kt+2
    for (int kt = 0; kt < KT - 2; ++kt) {
        stage_async<BM>(gA, ldkb, (kt + 2) * BK * 2, sAb + nx2 * A_TILE_BYTES, tid);
        stage_async<BN>(gB, ldkb, (kt + 2) * BK * 2, sBb + nx2 * B_TILE_BYTES, tid);
        compute(cur);
        wait_async_le6();     // kt+1's stage (two steps old) has landed
        __syncthreads();
        cur = (cur == 2) ? 0 : cur + 1;
        nx2 = (nx2 == 2) ? 0 : nx2 + 1;
    }
    // peeled pipeline drain: kt = KT-2, then KT-1 (no barrier after the last compute)
    compute(cur);
    wait_async_le0();
    __syncthreads();
    cur = (cur == 2) ? 0 : cur + 1;
    compute(cur);

    // Epilogue: VGPR r, lane L -> row = r + 8*(L/16), col = L%16
    const int col_base = blockIdx.x * BN + wNoff;
    const int row_base = blockIdx.y * BM + wMoff;
    #pragma unroll
    for (int ni = 0; ni < 4; ++ni) {
        const int col = col_base + ni * 16 + lrow;
        float bv = 0.0f;
        if constexpr (HAS_BIAS) bv = bias[col];
        #pragma unroll
        for (int mi = 0; mi < 4; ++mi) {
            const int row0 = row_base + mi * 16 + khalf * 8;
            #pragma unroll
            for (int r = 0; r < 8; ++r) {
                float v = acc[mi][ni][r] * scale + bv;
                if constexpr (RELU) v = fmaxf(v, 0.0f);
                const size_t idx = (size_t)(row0 + r) * N + col;
                if constexpr (OUT_F16) ((_Float16*)Cout)[idx] = (_Float16)v;
                else                   ((float*)Cout)[idx]    = v;
            }
        }
    }
}

// GL coefficients: w0=1, w_k = w_{k-1}*(k-1-alpha)/k  (double cumprod like numpy)
__global__ void gl_coeff_kernel(float* __restrict__ w, int n, float alpha) {
    if (blockIdx.x == 0 && threadIdx.x == 0) {
        double acc = 1.0;
        w[0] = 1.0f;
        for (int k = 1; k < n; ++k) {
            acc *= ((double)k - 1.0 - (double)alpha) / (double)k;
            w[k] = (float)acc;
        }
    }
}

// Bt[row][k] = w[row-k] for k<=row else 0, written directly in f16 ([N,K] row-major)
__global__ void toeplitz_f16_kernel(const float* __restrict__ w, _Float16* __restrict__ Bt, int n) {
    int k   = blockIdx.x * blockDim.x + threadIdx.x;
    int row = blockIdx.y;
    if (k < n) Bt[(size_t)row * n + k] = (k <= row) ? (_Float16)w[row - k] : (_Float16)0.0f;
}

// Bulk f32 -> f16 conversion (float4 granularity)
__global__ void cvt_f32_f16_kernel(const float* __restrict__ src, _Float16* __restrict__ dst, int n4) {
    int i = blockIdx.x * blockDim.x + threadIdx.x;
    if (i < n4) {
        f32x4 v = ((const f32x4*)src)[i];
        f16x4 hv = { (_Float16)v.x, (_Float16)v.y, (_Float16)v.z, (_Float16)v.w };
        ((f16x4*)dst)[i] = hv;
    }
}

extern "C" void kernel_launch(void* const* d_in, const int* in_sizes, int n_in,
                              void* d_out, int out_size, void* d_ws, size_t ws_size,
                              hipStream_t stream) {
    (void)in_sizes; (void)n_in; (void)out_size; (void)ws_size;
    const float* x  = (const float*)d_in[0];
    const float* W0 = (const float*)d_in[1];
    const float* b0 = (const float*)d_in[2];
    const float* W1 = (const float*)d_in[3];
    const float* b1 = (const float*)d_in[4];
    const float* W2 = (const float*)d_in[5];
    const float* b2 = (const float*)d_in[6];
    float* out = (float*)d_out;

    const int Bsz = 4096, N = 2048, H1 = 4096, H2 = 4096, DOUT = 1024;

    char* ws = (char*)d_ws;
    float*    wco = (float*)ws;                         size_t off = 64 * 1024;
    _Float16* Bt  = (_Float16*)(ws + off);              off += (size_t)N * N * 2;      // 8MB
    _Float16* x16 = (_Float16*)(ws + off);              off += (size_t)Bsz * N * 2;    // 16MB
    _Float16* W0h = (_Float16*)(ws + off);              off += (size_t)H1 * N * 2;     // 16MB
    _Float16* W1h = (_Float16*)(ws + off);              off += (size_t)H2 * H1 * 2;    // 32MB
    _Float16* W2h = (_Float16*)(ws + off);              off += (size_t)DOUT * H2 * 2;  // 8MB
    _Float16* h   = (_Float16*)(ws + off);              off += (size_t)Bsz * N * 2;    // 16MB
    _Float16* h0  = (_Float16*)(ws + off);              off += (size_t)Bsz * H1 * 2;   // 32MB
    _Float16* h1  = (_Float16*)(ws + off);              // 32MB

    gl_coeff_kernel<<<1, 32, 0, stream>>>(wco, N, 0.5f);
    toeplitz_f16_kernel<<<dim3(N / 256, N), 256, 0, stream>>>(wco, Bt, N);

    cvt_f32_f16_kernel<<<(Bsz * N / 4 + 255) / 256, 256, 0, stream>>>(x,  x16, Bsz * N / 4);
    cvt_f32_f16_kernel<<<(H1 * N / 4 + 255) / 256, 256, 0, stream>>>(W0, W0h, H1 * N / 4);
    cvt_f32_f16_kernel<<<(H2 * H1 / 4 + 255) / 256, 256, 0, stream>>>(W1, W1h, H2 * H1 / 4);
    cvt_f32_f16_kernel<<<(DOUT * H2 / 4 + 255) / 256, 256, 0, stream>>>(W2, W2h, DOUT * H2 / 4);

    const float scale = sqrtf((float)(N - 1));   // H^{-alpha} = (N-1)^0.5, alpha=0.5

    gemm_f16_kernel<false, false, true,  true ><<<dim3(N    / BN, Bsz / BM), NTHREADS, 0, stream>>>(x16, Bt,  nullptr, h,   Bsz, N,    N,  scale);
    gemm_f16_kernel<true,  true,  true,  false><<<dim3(H1   / BN, Bsz / BM), NTHREADS, 0, stream>>>(h,   W0h, b0,      h0,  Bsz, H1,   N,  1.0f);
    gemm_f16_kernel<true,  true,  true,  false><<<dim3(H2   / BN, Bsz / BM), NTHREADS, 0, stream>>>(h0,  W1h, b1,      h1,  Bsz, H2,   H1, 1.0f);
    gemm_f16_kernel<false, true,  false, false><<<dim3(DOUT / BN, Bsz / BM), NTHREADS, 0, stream>>>(h1,  W2h, b2,      out, Bsz, DOUT, H2, 1.0f);
}
